// MultiHeadAttentionBlock_26534307955412
// MI455X (gfx1250) — matrix operand order repaired and checked by reading, not compiled, not run on hardware
//
#include <hip/hip_runtime.h>

// ---------------------------------------------------------------------------
// MHA block on gfx1250: bf16 WMMA projections + async-LDS flash attention.
// ---------------------------------------------------------------------------

#define B_NUM  4
#define S_LEN  2048
#define D_DIM  1024
#define H_NUM  16
#define DK     64
#define MTOT   (B_NUM * S_LEN)   // 8192 rows for the projection GEMMs

typedef __attribute__((ext_vector_type(16))) __bf16 v16bf;
typedef __attribute__((ext_vector_type(8)))  float  v8f;

union Frag {           // one WMMA 16-bit A or B operand: 16 bf16 = 8 VGPRs
    v16bf v;
    uint4 u4[2];
};

// round-to-nearest-even fp32 -> bf16 (raw ushort)
static __device__ __forceinline__ unsigned short f2bf(float x) {
    unsigned int u = __float_as_uint(x);
    u += 0x7FFFu + ((u >> 16) & 1u);
    return (unsigned short)(u >> 16);
}

// wave-relative LDS byte offset of a __shared__ object
static __device__ __forceinline__ unsigned lds_off(const void* p) {
    return (unsigned)(unsigned long long)
        (__attribute__((address_space(3))) const void*)p;
}

// one 16-byte async global->LDS copy per lane (ASYNCcnt-tracked)
static __device__ __forceinline__ void async_copy16(unsigned lds_byte_off,
                                                    const unsigned short* gptr) {
    unsigned long long ga = (unsigned long long)(size_t)gptr;
    asm volatile("global_load_async_to_lds_b128 %0, %1, off"
                 :: "v"(lds_byte_off), "v"(ga) : "memory");
}

// ---------------------------------------------------------------------------
// Elementwise fp32 -> bf16 (vectorized by 4)
// ---------------------------------------------------------------------------
__global__ void cvt_f32_bf16(const float* __restrict__ in,
                             unsigned short* __restrict__ out, int n4) {
    int i = blockIdx.x * blockDim.x + threadIdx.x;
    if (i < n4) {
        float4 f = ((const float4*)in)[i];
        ushort4 o;
        o.x = f2bf(f.x); o.y = f2bf(f.y); o.z = f2bf(f.z); o.w = f2bf(f.w);
        ((ushort4*)out)[i] = o;
    }
}

// ---------------------------------------------------------------------------
// WMMA GEMM: out[m][n] = sum_k X[m][k] * W[n][k] + bias[n]
//   X: [M][K] bf16 row-major (A operand), W: [N][K] bf16 row-major (B operand)
//   4 waves / block, each wave computes a 16(M) x 64(N) tile.
//   Software-pipelined K loop (ping-pong fragment registers).
//   mode 0: bf16 [b][h][s][dk]   mode 1: bf16 [b][h][dk][s]   mode 2: fp32 [m][n]
// ---------------------------------------------------------------------------
static __device__ __forceinline__ void loadA(Frag& a, const unsigned short* arow,
                                             int k0, int halo) {
    a.u4[0] = *(const uint4*)(arow + k0 + halo * 8);
    a.u4[1] = *(const uint4*)(arow + k0 + 16 + halo * 8);
}
static __device__ __forceinline__ void loadB4(Frag* b, const unsigned short* W,
                                              int ntile, int l16, int halo,
                                              int K, int k0) {
    #pragma unroll
    for (int t = 0; t < 4; ++t) {
        const unsigned short* brow =
            W + (size_t)(ntile + t * 16 + l16) * K + k0 + halo * 8;
        b[t].u4[0] = *(const uint4*)(brow);
        b[t].u4[1] = *(const uint4*)(brow + 16);
    }
}

__global__ void proj_gemm(const unsigned short* __restrict__ X,
                          const unsigned short* __restrict__ W,
                          const float* __restrict__ bias,
                          void* __restrict__ out, int K, int mode) {
    const int lane  = threadIdx.x & 31;
    const int wave  = threadIdx.x >> 5;
    const int halo  = lane >> 4;    // 0/1 half-wave
    const int l16   = lane & 15;
    const int mtile = (blockIdx.x * 4 + wave) * 16;
    const int ntile = blockIdx.y * 64;

    v8f acc[4];
    #pragma unroll
    for (int t = 0; t < 4; ++t)
        #pragma unroll
        for (int r = 0; r < 8; ++r) acc[t][r] = 0.0f;

    const unsigned short* arow = X + (size_t)(mtile + l16) * K;

    Frag a0, a1, b0[4], b1[4];
    loadA(a0, arow, 0, halo);
    loadB4(b0, W, ntile, l16, halo, K, 0);

    for (int k0 = 0; k0 < K; k0 += 64) {
        __builtin_prefetch(arow + k0 + 128, 0, 0);
        // prefetch k0+32 while computing k0
        loadA(a1, arow, k0 + 32, halo);
        loadB4(b1, W, ntile, l16, halo, K, k0 + 32);
        #pragma unroll
        for (int t = 0; t < 4; ++t)
            acc[t] = __builtin_amdgcn_wmma_f32_16x16x32_bf16(
                false, a0.v, false, b0[t].v, (short)0, acc[t], false, false);
        // prefetch k0+64 while computing k0+32
        if (k0 + 64 < K) {
            loadA(a0, arow, k0 + 64, halo);
            loadB4(b0, W, ntile, l16, halo, K, k0 + 64);
        }
        #pragma unroll
        for (int t = 0; t < 4; ++t)
            acc[t] = __builtin_amdgcn_wmma_f32_16x16x32_bf16(
                false, a1.v, false, b1[t].v, (short)0, acc[t], false, false);
    }

    #pragma unroll
    for (int t = 0; t < 4; ++t) {
        const int n  = ntile + t * 16 + l16;
        const float bv = bias[n];
        #pragma unroll
        for (int r = 0; r < 8; ++r) {
            const int m = mtile + r + 8 * halo;
            const float val = acc[t][r] + bv;
            if (mode == 2) {
                ((float*)out)[(size_t)m * D_DIM + n] = val;
            } else {
                const int b  = m / S_LEN, s = m % S_LEN;
                const int h  = n / DK,   dk = n % DK;
                size_t idx;
                if (mode == 0)
                    idx = (((size_t)(b * H_NUM + h) * S_LEN + s) * DK + dk);
                else
                    idx = (((size_t)(b * H_NUM + h) * DK + dk) * S_LEN + s);
                ((unsigned short*)out)[idx] = f2bf(val);
            }
        }
    }
}

// ---------------------------------------------------------------------------
// Flash attention, 4 waves / WG over a 64-query supertile of one (b,h).
// K/V 32-key tiles are double-buffered in LDS via global_load_async_to_lds,
// pipelined with s_wait_asynccnt. Each wave owns 16 queries.
//   Qp/Kp: [b][h][s][dk] bf16, Vt: [b][h][dk][s] bf16
//   ctx out: [b][s][h*dk] bf16 (A-operand-ready for the output projection)
// ---------------------------------------------------------------------------
__global__ void flash_attn(const unsigned short* __restrict__ Qp,
                           const unsigned short* __restrict__ Kp,
                           const unsigned short* __restrict__ Vt,
                           const int* __restrict__ mask,
                           unsigned short* __restrict__ ctx) {
    const int tid   = threadIdx.x;
    const int wave  = tid >> 5;
    const int lane  = tid & 31;
    const int halo  = lane >> 4;
    const int l16   = lane & 15;
    const int qtile = blockIdx.x * 64 + wave * 16;
    const int h     = blockIdx.y;
    const int b     = blockIdx.z;

    __shared__ __align__(16) unsigned short sK[2][32][DK];   // [key][d]
    __shared__ __align__(16) unsigned short sV[2][DK][32];   // [d][key]
    __shared__ __align__(16) float          s_sc[4][16][32]; // per-wave scores
    __shared__ __align__(16) unsigned short s_p[4][16][32];  // per-wave P (bf16)
    __shared__ float s_alpha[4][16];
    __shared__ float s_inv[4][16];

    const size_t head_off = (size_t)(b * H_NUM + h) * S_LEN * DK;
    const unsigned short* Kbase0 = Kp + head_off;   // + key0*DK per tile
    const unsigned short* Vbase0 = Vt + head_off;   // + key0 per tile (row d: +d*S)

    // stage one 32-key K tile (4KB contiguous) + V tile (64 rows x 64B):
    // 256 16-byte chunks split over 128 threads; 4 async instrs per wave.
    auto stage = [&](int buf, int key0) {
        const unsigned short* Kb = Kbase0 + (size_t)key0 * DK;
        const unsigned short* Vb = Vbase0 + key0;
        const unsigned kofs = lds_off(&sK[buf][0][0]);
        const unsigned vofs = lds_off(&sV[buf][0][0]);
        #pragma unroll
        for (int i = 0; i < 2; ++i) {
            const int c = tid + i * 128;          // 16B chunk index 0..255
            async_copy16(kofs + c * 16, Kb + c * 8);
            const int d = c >> 2, seg = c & 3;
            async_copy16(vofs + c * 16, Vb + (size_t)d * S_LEN + seg * 8);
        }
    };

    // Q tile as two A fragments (d = 0..31, 32..63), loaded once
    Frag aq[2];
    {
        const unsigned short* qrow = Qp + head_off + (size_t)(qtile + l16) * DK;
        #pragma unroll
        for (int c = 0; c < 2; ++c) {
            aq[c].u4[0] = *(const uint4*)(qrow + c * 32 + halo * 8);
            aq[c].u4[1] = *(const uint4*)(qrow + c * 32 + 16 + halo * 8);
        }
    }

    v8f acc[4];
    #pragma unroll
    for (int t = 0; t < 4; ++t)
        #pragma unroll
        for (int r = 0; r < 8; ++r) acc[t][r] = 0.0f;
    float m_i = -3.0e38f, l_i = 0.0f;

    stage(0, 0);                                   // prologue: tile 0 in flight

    const int NT = S_LEN / 32;                     // 64 key tiles
    for (int it = 0; it < NT; ++it) {
        const int key0 = it * 32;
        const int cur  = it & 1;

        if (it + 1 < NT) {
            stage((it + 1) & 1, key0 + 32);        // keep next tile in flight
            asm volatile("s_wait_asynccnt 0x4" ::: "memory");  // older batch done
        } else {
            asm volatile("s_wait_asynccnt 0x0" ::: "memory");
        }
        __syncthreads();                           // publish all waves' copies

        // ---- scores 16q x 32k from LDS K tile
        #pragma unroll
        for (int j = 0; j < 2; ++j) {
            v8f sc;
            #pragma unroll
            for (int r = 0; r < 8; ++r) sc[r] = 0.0f;
            #pragma unroll
            for (int c = 0; c < 2; ++c) {
                const unsigned short* kr = &sK[cur][j * 16 + l16][c * 32 + halo * 8];
                Frag bk;
                bk.u4[0] = *(const uint4*)(kr);
                bk.u4[1] = *(const uint4*)(kr + 16);
                sc = __builtin_amdgcn_wmma_f32_16x16x32_bf16(
                    false, aq[c].v, false, bk.v, (short)0, sc, false, false);
            }
            #pragma unroll
            for (int r = 0; r < 8; ++r)
                s_sc[wave][r + 8 * halo][j * 16 + l16] = sc[r] * 0.125f; // 1/sqrt(DK)
        }

        // ---- online softmax: lane L < 16 owns query row L of this wave
        if (lane < 16) {
            const int* mrow = mask + (size_t)(qtile + lane) * S_LEN + key0;
            float sv[32];
            float tmax = -3.0e38f;
            #pragma unroll
            for (int j = 0; j < 32; ++j) {
                float vv = s_sc[wave][lane][j];
                vv = (mrow[j] == 0) ? -1.0e9f : vv;
                sv[j] = vv;
                tmax = fmaxf(tmax, vv);
            }
            const float new_m = fmaxf(m_i, tmax);
            const float alpha = __expf(m_i - new_m);
            float sum = 0.0f;
            #pragma unroll
            for (int j = 0; j < 32; ++j) {
                float p = __expf(sv[j] - new_m);
                sum += p;
                s_p[wave][lane][j] = f2bf(p);
            }
            l_i = l_i * alpha + sum;
            m_i = new_m;
            s_alpha[wave][lane] = alpha;
        }

        // ---- rescale running accumulators by per-row alpha
        float al[8];
        #pragma unroll
        for (int r = 0; r < 8; ++r) al[r] = s_alpha[wave][r + 8 * halo];
        #pragma unroll
        for (int t = 0; t < 4; ++t)
            #pragma unroll
            for (int r = 0; r < 8; ++r) acc[t][r] *= al[r];

        // ---- P (16x32 bf16) from this wave's LDS slice as A fragment
        Frag ap;
        ap.u4[0] = *(const uint4*)(&s_p[wave][l16][halo * 8]);
        ap.u4[1] = *(const uint4*)(&s_p[wave][l16][16 + halo * 8]);

        // ---- acc += P x V from LDS V tile (4 d-blocks of 16)
        #pragma unroll
        for (int t = 0; t < 4; ++t) {
            const unsigned short* vr = &sV[cur][t * 16 + l16][halo * 8];
            Frag bv;
            bv.u4[0] = *(const uint4*)(vr);
            bv.u4[1] = *(const uint4*)(vr + 16);
            acc[t] = __builtin_amdgcn_wmma_f32_16x16x32_bf16(
                false, ap.v, false, bv.v, (short)0, acc[t], false, false);
        }
        __syncthreads();   // all waves done with buffer `cur` before reuse
    }

    if (lane < 16) s_inv[wave][lane] = 1.0f / l_i;
    float inv[8];
    #pragma unroll
    for (int r = 0; r < 8; ++r) inv[r] = s_inv[wave][r + 8 * halo];

    #pragma unroll
    for (int t = 0; t < 4; ++t)
        #pragma unroll
        for (int r = 0; r < 8; ++r) {
            const int s = qtile + r + 8 * halo;
            const int d = h * DK + t * 16 + l16;
            ctx[(size_t)(b * S_LEN + s) * D_DIM + d] = f2bf(acc[t][r] * inv[r]);
        }
}

// ---------------------------------------------------------------------------
// Host-side orchestration
// ---------------------------------------------------------------------------
extern "C" void kernel_launch(void* const* d_in, const int* in_sizes, int n_in,
                              void* d_out, int out_size, void* d_ws, size_t ws_size,
                              hipStream_t stream) {
    (void)in_sizes; (void)n_in; (void)out_size; (void)ws_size;

    const float* q    = (const float*)d_in[0];
    const float* k    = (const float*)d_in[1];
    const float* v    = (const float*)d_in[2];
    const int*   mask = (const int*)  d_in[3];
    const float* w_q  = (const float*)d_in[4];
    const float* b_q  = (const float*)d_in[5];
    const float* w_k  = (const float*)d_in[6];
    const float* b_k  = (const float*)d_in[7];
    const float* w_v  = (const float*)d_in[8];
    const float* b_v  = (const float*)d_in[9];
    const float* w_o  = (const float*)d_in[10];
    const float* b_o  = (const float*)d_in[11];

    // workspace carve-up (bf16 buffers)
    const size_t xelem = (size_t)MTOT * D_DIM;        // 8M elements
    const size_t welem = (size_t)D_DIM * D_DIM;       // 1M elements
    char* p = (char*)d_ws;
    unsigned short* xq  = (unsigned short*)p; p += xelem * 2;
    unsigned short* xk  = (unsigned short*)p; p += xelem * 2;
    unsigned short* xv  = (unsigned short*)p; p += xelem * 2;
    unsigned short* wqb = (unsigned short*)p; p += welem * 2;
    unsigned short* wkb = (unsigned short*)p; p += welem * 2;
    unsigned short* wvb = (unsigned short*)p; p += welem * 2;
    unsigned short* wob = (unsigned short*)p; p += welem * 2;
    unsigned short* Qp  = (unsigned short*)p; p += xelem * 2;
    unsigned short* Kp  = (unsigned short*)p; p += xelem * 2;
    unsigned short* Vt  = (unsigned short*)p; p += xelem * 2;
    unsigned short* ctx = (unsigned short*)p; p += xelem * 2;

    // 1) fp32 -> bf16 conversions
    {
        const int n4x = (int)(xelem / 4);
        const int n4w = (int)(welem / 4);
        dim3 bk256(256);
        cvt_f32_bf16<<<dim3((n4x + 255) / 256), bk256, 0, stream>>>(q, xq, n4x);
        cvt_f32_bf16<<<dim3((n4x + 255) / 256), bk256, 0, stream>>>(k, xk, n4x);
        cvt_f32_bf16<<<dim3((n4x + 255) / 256), bk256, 0, stream>>>(v, xv, n4x);
        cvt_f32_bf16<<<dim3((n4w + 255) / 256), bk256, 0, stream>>>(w_q, wqb, n4w);
        cvt_f32_bf16<<<dim3((n4w + 255) / 256), bk256, 0, stream>>>(w_k, wkb, n4w);
        cvt_f32_bf16<<<dim3((n4w + 255) / 256), bk256, 0, stream>>>(w_v, wvb, n4w);
        cvt_f32_bf16<<<dim3((n4w + 255) / 256), bk256, 0, stream>>>(w_o, wob, n4w);
    }

    // 2) Q/K/V projections (WMMA)
    dim3 gg(MTOT / 64, D_DIM / 64);   // 128 x 16 blocks, 4 waves each
    dim3 gb(128);
    proj_gemm<<<gg, gb, 0, stream>>>(xq, wqb, b_q, (void*)Qp, D_DIM, 0);
    proj_gemm<<<gg, gb, 0, stream>>>(xk, wkb, b_k, (void*)Kp, D_DIM, 0);
    proj_gemm<<<gg, gb, 0, stream>>>(xv, wvb, b_v, (void*)Vt, D_DIM, 1);

    // 3) flash attention (4 waves per WG share async-staged K/V tiles)
    flash_attn<<<dim3(S_LEN / 64, H_NUM, B_NUM), dim3(128), 0, stream>>>(
        Qp, Kp, Vt, mask, ctx);

    // 4) output projection (fp32 result)
    proj_gemm<<<gg, gb, 0, stream>>>(ctx, wob, b_o, d_out, D_DIM, 2);
}
